// scope_net_attention_80333068304423
// MI455X (gfx1250) — compile-verified, matrix operand order checked by reading, not run on hardware
//
#include <hip/hip_runtime.h>
#include <hip/hip_bf16.h>

// ---------------------------------------------------------------------------
// scope_net_attention for MI455X (gfx1250), compile-only tuning.
//
// 4-pass recompute scheme (BatchNorm training-mode needs batch stats):
//   setup   : zero stat accumulators, repack W2/dW1/dW2 into f16 WMMA-B layout
//   pass1   : stats (sum/sumsq) of k/q/v pre-BN activations  [3 x 32 feats]
//   finalize: mean/var -> (scale, shift) per feature
//   pass2   : recompute -> k,q,v GEMMs (WMMA) -> softmax -> attn@dW1 stats
//   pass3   : recompute -> BN -> h@dW2 stats
//   pass4   : recompute -> final 16->1 + sigmoid -> out
// One wave handles a 16-row tile; all GEMMs are v_wmma_f32_16x16x32_f16.
// ---------------------------------------------------------------------------

#define N_ROWS 2097152
#define EPSBN  1e-5f

typedef __attribute__((ext_vector_type(16))) _Float16 v16h;
typedef __attribute__((ext_vector_type(8)))  _Float16 v8h;
typedef __attribute__((ext_vector_type(8)))  float    v8f;

// ---- workspace layout (float offsets) ----
#define ST_K   0      // sum[32], sumsq[32]
#define ST_Q   64
#define ST_V   128
#define ST_D1  192
#define ST_D2  256    // sum[16], sumsq[16]
#define STATS_FLOATS 288
#define SS_K   288    // scale[32], shift[32]
#define SS_Q   352
#define SS_V   416
#define SS_D1  480
#define SS_D2  544    // scale[16], shift[16]
// packed f16 B-operands start at byte 4096 of ws (half offsets below)
#define HB_K   0
#define HB_Q   2048
#define HB_V   4096
#define HB_D1  6144
#define HB_D2  8192
#define HB_TOTAL 8704

// ---------------------------------------------------------------------------
__device__ inline v8f wmma16(v16h a, v16h b, v8f c) {
  // (neg_a, A, neg_b, B, c_mod, C, reuse_a, reuse_b)
  return __builtin_amdgcn_wmma_f32_16x16x32_f16(false, a, false, b,
                                                (short)0, c, false, false);
}

__device__ inline v16h loadB(const _Float16* base, int tile, int lane) {
  return *reinterpret_cast<const v16h*>(base + tile * 512 + lane * 16);
}

__device__ inline v16h hcat(v8h lo, v8h hi) {
  v16h r;
#pragma unroll
  for (int i = 0; i < 8; ++i) { r[i] = lo[i]; r[8 + i] = hi[i]; }
  return r;
}

// Compute 16 BN'd+ReLU'd features of this lane's row, directly in the
// f16 A-operand layout: lane L holds row M=L%16; halves 0..7 -> K=kLo..kLo+7,
// halves 8..15 -> K=16+kLo..16+kLo+7 (kLo = 0 for lanes 0-15, 8 for 16-31).
template <int K>
__device__ inline v16h lane_features(const float* xr, const float* W,
                                     const float* ss, int kLo) {
  v16h a;
#pragma unroll
  for (int i = 0; i < 16; ++i) {
    const int f = (i < 8) ? (kLo + i) : (kLo + i + 8);
    float acc = 0.f;
#pragma unroll
    for (int k = 0; k < K; ++k) acc = fmaf(xr[k], W[k * 32 + f], acc);
    float y = fmaxf(fmaf(acc, ss[f], ss[32 + f]), 0.f);
    a[i] = (_Float16)y;
  }
  return a;
}

struct SnaP {
  const float *rot, *tra;
  const float *kW1, *qW1, *vW1;
  const float *kb2, *qb2, *vb2;
  const float *dW3, *db3;
  const float *wsF;        // stats + scale/shift region
  const _Float16 *wsH;     // packed f16 B operands
  float *stats;            // per-pass global stat accumulators
  float *out;
};

// k,q,v 32->64 GEMMs + softmax(k*q)*v ; attn left in C-layout (lane=col).
__device__ inline void sna_attn(const SnaP& p, int rowBase, int lane,
                                float attn[4][8]) {
  const int m   = lane & 15;
  const int kLo = (lane & 16) ? 8 : 0;
  const long row = rowBase + m;
  float tr[3], ro[9];
#pragma unroll
  for (int i = 0; i < 3; ++i) tr[i] = p.tra[row * 3 + i];
#pragma unroll
  for (int i = 0; i < 9; ++i) ro[i] = p.rot[row * 9 + i];

  v16h aK = lane_features<3>(tr, p.kW1, p.wsF + SS_K, kLo);
  v16h aQ = lane_features<9>(ro, p.qW1, p.wsF + SS_Q, kLo);
  v16h aV = lane_features<3>(tr, p.vW1, p.wsF + SS_V, kLo);

  float s[4][8], vv[4][8];
#pragma unroll
  for (int t = 0; t < 4; ++t) {
    v8f z = {};
    v8f cK = wmma16(aK, loadB(p.wsH + HB_K, t, lane), z);
    v8f cQ = wmma16(aQ, loadB(p.wsH + HB_Q, t, lane), z);
    v8f cV = wmma16(aV, loadB(p.wsH + HB_V, t, lane), z);
    const int col = t * 16 + m;
    const float bk = p.kb2[col], bq = p.qb2[col], bv = p.vb2[col];
#pragma unroll
    for (int j = 0; j < 8; ++j) {
      s[t][j]  = (cK[j] + bk) * (cQ[j] + bq);
      vv[t][j] = cV[j] + bv;
    }
  }
  // softmax over the 64 columns of each row; columns live across the 16
  // lanes of each half-wave (xor masks 1,2,4,8 stay inside a half).
#pragma unroll
  for (int j = 0; j < 8; ++j) {
    float mx = fmaxf(fmaxf(s[0][j], s[1][j]), fmaxf(s[2][j], s[3][j]));
#pragma unroll
    for (int d = 1; d < 16; d <<= 1) mx = fmaxf(mx, __shfl_xor(mx, d, 32));
    float e0 = __expf(s[0][j] - mx), e1 = __expf(s[1][j] - mx);
    float e2 = __expf(s[2][j] - mx), e3 = __expf(s[3][j] - mx);
    float sm = e0 + e1 + e2 + e3;
#pragma unroll
    for (int d = 1; d < 16; d <<= 1) sm += __shfl_xor(sm, d, 32);
    const float inv = 1.f / sm;
    attn[0][j] = e0 * inv * vv[0][j];
    attn[1][j] = e1 * inv * vv[1][j];
    attn[2][j] = e2 * inv * vv[2][j];
    attn[3][j] = e3 * inv * vv[3][j];
  }
}

// a1 = attn(16x64) @ dW1(64x32) via LDS transpose (C-layout -> A-layout).
__device__ inline void sna_a1(const SnaP& p, const float attn[4][8], int lane,
                              _Float16* ldsW, float a1[2][8]) {
  const int m    = lane & 15;
  const int kLo  = (lane & 16) ? 8 : 0;
  const int rOff = (lane & 16) ? 8 : 0;
#pragma unroll
  for (int t = 0; t < 4; ++t)
#pragma unroll
    for (int j = 0; j < 8; ++j)
      ldsW[(j + rOff) * 72 + t * 16 + m] = (_Float16)attn[t][j];
  asm volatile("s_wait_dscnt 0x0" ::: "memory");  // same-wave LDS RAW fence
  const _Float16* rowp = ldsW + m * 72;
  v8h g0 = *(const v8h*)(rowp + kLo);
  v8h g1 = *(const v8h*)(rowp + 16 + kLo);
  v8h g2 = *(const v8h*)(rowp + 32 + kLo);
  v8h g3 = *(const v8h*)(rowp + 48 + kLo);
  v16h aLo = hcat(g0, g1), aHi = hcat(g2, g3);
#pragma unroll
  for (int nt = 0; nt < 2; ++nt) {
    v8f z = {};
    v8f c = wmma16(aLo, loadB(p.wsH + HB_D1, nt, lane), z);
    c     = wmma16(aHi, loadB(p.wsH + HB_D1, 2 + nt, lane), c);
#pragma unroll
    for (int j = 0; j < 8; ++j) a1[nt][j] = c[j];
  }
}

// h2 = relu(BN(a1))(16x32) @ dW2(32x16), same LDS transpose trick.
__device__ inline void sna_h2(const SnaP& p, const float a1[2][8], int lane,
                              _Float16* ldsW, float h2[8]) {
  const int m    = lane & 15;
  const int kLo  = (lane & 16) ? 8 : 0;
  const int rOff = (lane & 16) ? 8 : 0;
  const float* ss = p.wsF + SS_D1;
#pragma unroll
  for (int nt = 0; nt < 2; ++nt) {
    const int col = nt * 16 + m;
    const float sc = ss[col], sh = ss[32 + col];
#pragma unroll
    for (int j = 0; j < 8; ++j) {
      float y = fmaxf(fmaf(a1[nt][j], sc, sh), 0.f);
      ldsW[(j + rOff) * 40 + col] = (_Float16)y;
    }
  }
  asm volatile("s_wait_dscnt 0x0" ::: "memory");
  const _Float16* rowp = ldsW + m * 40;
  v8h g0 = *(const v8h*)(rowp + kLo);
  v8h g1 = *(const v8h*)(rowp + 16 + kLo);
  v16h a = hcat(g0, g1);
  v8f z = {};
  v8f c = wmma16(a, loadB(p.wsH + HB_D2, 0, lane), z);
#pragma unroll
  for (int j = 0; j < 8; ++j) h2[j] = c[j];
}

// ---------------------------------------------------------------------------
// setup: zero stat accumulators, repack B operands to f16 WMMA layout.
// B tile (32K x 16N): lane L holds col N=L%16; halves mirror the A layout.
__global__ void sna_setup(const float* kW2, const float* qW2, const float* vW2,
                          const float* dW1, const float* dW2,
                          float* wsF, _Float16* wsH) {
  for (int i = threadIdx.x; i < STATS_FLOATS; i += blockDim.x) wsF[i] = 0.f;
  for (int e = threadIdx.x; e < HB_TOTAL; e += blockDim.x) {
    int r, ld, k0, c0; const float* W;
    if (e < 2048)      { r = e;        W = kW2; ld = 64; int t = r >> 9; k0 = 0;            c0 = t * 16; }
    else if (e < 4096) { r = e - 2048; W = qW2; ld = 64; int t = r >> 9; k0 = 0;            c0 = t * 16; }
    else if (e < 6144) { r = e - 4096; W = vW2; ld = 64; int t = r >> 9; k0 = 0;            c0 = t * 16; }
    else if (e < 8192) { r = e - 6144; W = dW1; ld = 32; int t = r >> 9; k0 = (t >> 1) * 32; c0 = (t & 1) * 16; }
    else               { r = e - 8192; W = dW2; ld = 16; k0 = 0;                            c0 = 0; }
    const int within = r & 511;
    const int lane = within >> 4;
    const int h    = within & 15;
    const int kLo  = (lane < 16) ? 0 : 8;
    const int k    = k0 + ((h < 8) ? (kLo + h) : (kLo + h + 8));
    const int c    = c0 + (lane & 15);
    wsH[e] = (_Float16)W[k * ld + c];
  }
}

// ---------------------------------------------------------------------------
// pass1: batch stats of the three pre-BN input projections.
template <int K>
__device__ void sna_acc_net(const float* x, const float* W, float* sacc,
                            int tid, int stride) {
  float sum[32], sq[32];
#pragma unroll
  for (int f = 0; f < 32; ++f) { sum[f] = 0.f; sq[f] = 0.f; }
  for (long r = tid; r < N_ROWS; r += stride) {
    float xr[K];
#pragma unroll
    for (int k = 0; k < K; ++k) xr[k] = x[r * K + k];
#pragma unroll
    for (int f = 0; f < 32; ++f) {
      float a = 0.f;
#pragma unroll
      for (int k = 0; k < K; ++k) a = fmaf(xr[k], W[k * 32 + f], a);
      sum[f] += a; sq[f] += a * a;
    }
  }
#pragma unroll
  for (int f = 0; f < 32; ++f) {
    float s = sum[f], q = sq[f];
#pragma unroll
    for (int d = 1; d < 32; d <<= 1) { s += __shfl_xor(s, d, 32); q += __shfl_xor(q, d, 32); }
    if ((threadIdx.x & 31) == 0) { atomicAdd(&sacc[f], s); atomicAdd(&sacc[32 + f], q); }
  }
}

__global__ void __launch_bounds__(256)
sna_pass1(const float* rot, const float* tra, const float* kW1,
          const float* qW1, const float* vW1, float* ws) {
  __shared__ float sacc[192];
  for (int i = threadIdx.x; i < 192; i += blockDim.x) sacc[i] = 0.f;
  __syncthreads();
  const int tid = blockIdx.x * blockDim.x + threadIdx.x;
  const int stride = gridDim.x * blockDim.x;
  sna_acc_net<3>(tra, kW1, sacc + ST_K, tid, stride);
  sna_acc_net<9>(rot, qW1, sacc + ST_Q, tid, stride);
  sna_acc_net<3>(tra, vW1, sacc + ST_V, tid, stride);
  __syncthreads();
  for (int i = threadIdx.x; i < 192; i += blockDim.x) atomicAdd(&ws[i], sacc[i]);
}

__global__ void sna_finalize(float* ws, int statOff, int ssOff,
                             const float* gamma, const float* beta, int nfeat) {
  const int f = threadIdx.x;
  if (f >= nfeat) return;
  const float invN = 1.f / (float)N_ROWS;
  const float mean = ws[statOff + f] * invN;
  const float var  = ws[statOff + nfeat + f] * invN - mean * mean;
  const float sc   = gamma[f] * rsqrtf(var + EPSBN);
  ws[ssOff + f] = sc;
  ws[ssOff + nfeat + f] = fmaf(-mean, sc, beta[f]);
}

// ---------------------------------------------------------------------------
__global__ void __launch_bounds__(256) sna_pass2(SnaP p) {
  __shared__ __align__(32) _Float16 ldsT[8][16 * 72];
  __shared__ float acc[64];
  for (int i = threadIdx.x; i < 64; i += blockDim.x) acc[i] = 0.f;
  __syncthreads();
  const int wave = threadIdx.x >> 5, lane = threadIdx.x & 31;
  const int rowBase = (blockIdx.x * 8 + wave) * 16;
  float attn[4][8]; sna_attn(p, rowBase, lane, attn);
  float a1[2][8];   sna_a1(p, attn, lane, ldsT[wave], a1);
#pragma unroll
  for (int nt = 0; nt < 2; ++nt) {
    float s = 0.f, q = 0.f;
#pragma unroll
    for (int j = 0; j < 8; ++j) { s += a1[nt][j]; q += a1[nt][j] * a1[nt][j]; }
    s += __shfl_xor(s, 16, 32); q += __shfl_xor(q, 16, 32);
    if (lane < 16) {
      const int col = nt * 16 + lane;
      atomicAdd(&acc[col], s); atomicAdd(&acc[32 + col], q);
    }
  }
  __syncthreads();
  for (int i = threadIdx.x; i < 64; i += blockDim.x) atomicAdd(&p.stats[i], acc[i]);
}

__global__ void __launch_bounds__(256) sna_pass3(SnaP p) {
  __shared__ __align__(32) _Float16 ldsT[8][16 * 72];
  __shared__ float acc[32];
  for (int i = threadIdx.x; i < 32; i += blockDim.x) acc[i] = 0.f;
  __syncthreads();
  const int wave = threadIdx.x >> 5, lane = threadIdx.x & 31;
  const int rowBase = (blockIdx.x * 8 + wave) * 16;
  float attn[4][8]; sna_attn(p, rowBase, lane, attn);
  float a1[2][8];   sna_a1(p, attn, lane, ldsT[wave], a1);
  float h2[8];      sna_h2(p, a1, lane, ldsT[wave], h2);
  float s = 0.f, q = 0.f;
#pragma unroll
  for (int j = 0; j < 8; ++j) { s += h2[j]; q += h2[j] * h2[j]; }
  s += __shfl_xor(s, 16, 32); q += __shfl_xor(q, 16, 32);
  if (lane < 16) { atomicAdd(&acc[lane], s); atomicAdd(&acc[16 + lane], q); }
  __syncthreads();
  for (int i = threadIdx.x; i < 32; i += blockDim.x) atomicAdd(&p.stats[i], acc[i]);
}

__global__ void __launch_bounds__(256) sna_pass4(SnaP p) {
  __shared__ __align__(32) _Float16 ldsT[8][16 * 72];
  const int wave = threadIdx.x >> 5, lane = threadIdx.x & 31;
  const int rowBase = (blockIdx.x * 8 + wave) * 16;
  float attn[4][8]; sna_attn(p, rowBase, lane, attn);
  float a1[2][8];   sna_a1(p, attn, lane, ldsT[wave], a1);
  float h2[8];      sna_h2(p, a1, lane, ldsT[wave], h2);
  const int m    = lane & 15;
  const int rOff = (lane & 16) ? 8 : 0;
  const float* ss = p.wsF + SS_D2;
  const float sc = ss[m], sh = ss[16 + m];
  const float w3 = p.dW3[m];
  const float b3 = p.db3[0];
#pragma unroll
  for (int j = 0; j < 8; ++j) {
    float y = fmaxf(fmaf(h2[j], sc, sh), 0.f) * w3;
#pragma unroll
    for (int d = 1; d < 16; d <<= 1) y += __shfl_xor(y, d, 32);
    if (m == 0) p.out[rowBase + rOff + j] = 1.f / (1.f + __expf(-(y + b3)));
  }
}

// ---------------------------------------------------------------------------
extern "C" void kernel_launch(void* const* d_in, const int* in_sizes, int n_in,
                              void* d_out, int out_size, void* d_ws, size_t ws_size,
                              hipStream_t stream) {
  (void)in_sizes; (void)n_in; (void)out_size; (void)ws_size;
  const float* rot = (const float*)d_in[0];
  const float* tra = (const float*)d_in[1];
  const float* kW1 = (const float*)d_in[2];
  const float* kg  = (const float*)d_in[3];
  const float* kb  = (const float*)d_in[4];
  const float* kW2 = (const float*)d_in[5];
  const float* kb2 = (const float*)d_in[6];
  const float* qW1 = (const float*)d_in[7];
  const float* qg  = (const float*)d_in[8];
  const float* qb  = (const float*)d_in[9];
  const float* qW2 = (const float*)d_in[10];
  const float* qb2 = (const float*)d_in[11];
  const float* vW1 = (const float*)d_in[12];
  const float* vg  = (const float*)d_in[13];
  const float* vb  = (const float*)d_in[14];
  const float* vW2 = (const float*)d_in[15];
  const float* vb2 = (const float*)d_in[16];
  const float* dW1 = (const float*)d_in[17];
  const float* dg1 = (const float*)d_in[18];
  const float* db1 = (const float*)d_in[19];
  const float* dW2 = (const float*)d_in[20];
  const float* dg2 = (const float*)d_in[21];
  const float* db2 = (const float*)d_in[22];
  const float* dW3 = (const float*)d_in[23];
  const float* db3 = (const float*)d_in[24];

  float* ws = (float*)d_ws;
  _Float16* wsH = (_Float16*)((char*)d_ws + 4096);

  sna_setup<<<1, 256, 0, stream>>>(kW2, qW2, vW2, dW1, dW2, ws, wsH);
  sna_pass1<<<2048, 256, 0, stream>>>(rot, tra, kW1, qW1, vW1, ws);
  sna_finalize<<<1, 32, 0, stream>>>(ws, ST_K, SS_K, kg, kb, 32);
  sna_finalize<<<1, 32, 0, stream>>>(ws, ST_Q, SS_Q, qg, qb, 32);
  sna_finalize<<<1, 32, 0, stream>>>(ws, ST_V, SS_V, vg, vb, 32);

  SnaP p;
  p.rot = rot; p.tra = tra;
  p.kW1 = kW1; p.qW1 = qW1; p.vW1 = vW1;
  p.kb2 = kb2; p.qb2 = qb2; p.vb2 = vb2;
  p.dW3 = dW3; p.db3 = db3;
  p.wsF = ws;  p.wsH = wsH;
  p.out = (float*)d_out;

  const int blocks = N_ROWS / 128;  // 8 waves x 16 rows per block
  p.stats = ws + ST_D1;
  sna_pass2<<<blocks, 256, 0, stream>>>(p);
  sna_finalize<<<1, 32, 0, stream>>>(ws, ST_D1, SS_D1, dg1, db1, 32);
  p.stats = ws + ST_D2;
  sna_pass3<<<blocks, 256, 0, stream>>>(p);
  sna_finalize<<<1, 16, 0, stream>>>(ws, ST_D2, SS_D2, dg2, db2, 16);
  sna_pass4<<<blocks, 256, 0, stream>>>(p);
}
